// STFT_72318659330195
// MI455X (gfx1250) — matrix operand order repaired and checked by reading, not compiled, not run on hardware
//
#include <hip/hip_runtime.h>
#include <math.h>

typedef __attribute__((ext_vector_type(16))) _Float16 v16h;
typedef __attribute__((ext_vector_type(8)))  _Float16 v8h;
typedef __attribute__((ext_vector_type(8)))  float    v8f;

#define FILTER_LEN 800
#define HOP        200
#define CUTOFF     401          // fl/2 + 1
#define NSAMP      8000000
#define NFRAMES    40001        // NSAMP/HOP + 1
#define KCHUNKS    25           // 800 / 32
#define MTILES     26           // ceil(401/16)
#define NTILES     2501         // ceil(40001/16)

union AFrag { v16h v; v8h h[2]; };

// One wave computes a 16x16 output tile: 16 freq bins x 16 frames, accumulating
// both real and imag parts with split-f16 (hi/lo) WMMA for ~fp32 accuracy.
// 4 waves per workgroup share a basis tile (16 real + 16 imag rows) in LDS.
__global__ __launch_bounds__(128) void stft_mag_wmma(
    const float* __restrict__ input,
    const float* __restrict__ basis,
    float* __restrict__ out)
{
    extern __shared__ _Float16 smem[];
    _Float16* sbH = smem;                      // [32][800] f16 hi
    _Float16* sbL = smem + 32 * FILTER_LEN;    // [32][800] f16 lo

    const int mtile = blockIdx.x;              // freq-tile: rows mtile*16 ..
    const int tid   = threadIdx.x;
    const int lane  = tid & 31;
    const int wave  = tid >> 5;

    // ---- stage basis tile into LDS as split f16 (rows 0..15 real, 16..31 imag)
    {
        int row_local = tid >> 2;              // 0..31 (4 threads per row)
        int kbase     = (tid & 3) * 200;       // 200 consecutive k per thread
        int m = mtile * 16 + (row_local & 15);
        if (m > CUTOFF - 1) m = CUTOFF - 1;    // clamp ragged M edge (store masked later)
        int grow = (row_local < 16) ? m : (m + CUTOFF);
        const float* src = basis + grow * FILTER_LEN + kbase;
        _Float16* dH = sbH + row_local * FILTER_LEN + kbase;
        _Float16* dL = sbL + row_local * FILTER_LEN + kbase;
        for (int k = 0; k < 200; ++k) {
            float v = src[k];
            _Float16 hi = (_Float16)v;
            dH[k] = hi;
            dL[k] = (_Float16)(v - (float)hi);
        }
    }
    __syncthreads();

    // ---- this wave's frame tile (wave-uniform control from here on)
    const int ntile = blockIdx.y * 4 + wave;
    if (ntile >= NTILES) return;               // whole wave exits together
    const int f0 = ntile * 16;

    int frame = f0 + (lane & 15);
    if (frame > NFRAMES - 1) frame = NFRAMES - 1;   // clamp loads; store masked
    // interior: every accessed padded index maps in-range without reflection
    const bool interior = (f0 >= 2) && (f0 + 15 <= 39998);

    const int rsel  = (lane >> 4) ? 8 : 0;     // A-layout lane-half K select
    const int rowR  = (lane & 15);             // A real row in LDS
    const int rowI  = rowR + 16;               // A imag row in LDS
    const int pbase = frame * HOP + (lane >> 4) * 8;   // B-layout lane-half K select

    v8f cr = {};                               // real accumulator (fp32)
    v8f ci = {};                               // imag accumulator (fp32)

    for (int c = 0; c < KCHUNKS; ++c) {
        // ---- B fragment: K samples (8h+s, 16+8h+s) for column `frame`
        float xs[16];
        const int p0 = pbase + c * 32;         // padded-domain index of first sample
        if (interior) {
            const float4* src = reinterpret_cast<const float4*>(input + (p0 - 400));
            float4 q0 = src[0], q1 = src[1], q2 = src[4], q3 = src[5];
            xs[0]=q0.x;  xs[1]=q0.y;  xs[2]=q0.z;  xs[3]=q0.w;
            xs[4]=q1.x;  xs[5]=q1.y;  xs[6]=q1.z;  xs[7]=q1.w;
            xs[8]=q2.x;  xs[9]=q2.y;  xs[10]=q2.z; xs[11]=q2.w;
            xs[12]=q3.x; xs[13]=q3.y; xs[14]=q3.z; xs[15]=q3.w;
        } else {
            #pragma unroll
            for (int i = 0; i < 16; ++i) {
                int s = p0 + i + (i >= 8 ? 8 : 0) - 400;   // reflect padding
                s = (s < 0) ? -s : s;
                if (s >= NSAMP) s = 2 * NSAMP - 2 - s;
                xs[i] = input[s];
            }
        }
        v16h bh, bl;
        #pragma unroll
        for (int i = 0; i < 16; ++i) {
            _Float16 hi = (_Float16)xs[i];
            bh[i] = hi;
            bl[i] = (_Float16)(xs[i] - (float)hi);
        }

        const int abase = c * 32 + rsel;

        // ---- real rows: C += Ahi*Bhi + Ahi*Blo + Alo*Bhi
        AFrag aH, aL;
        aH.h[0] = *reinterpret_cast<const v8h*>(sbH + rowR * FILTER_LEN + abase);
        aH.h[1] = *reinterpret_cast<const v8h*>(sbH + rowR * FILTER_LEN + abase + 16);
        aL.h[0] = *reinterpret_cast<const v8h*>(sbL + rowR * FILTER_LEN + abase);
        aL.h[1] = *reinterpret_cast<const v8h*>(sbL + rowR * FILTER_LEN + abase + 16);
        cr = __builtin_amdgcn_wmma_f32_16x16x32_f16(false, aH.v, false, bh, (short)0, cr, false, false);
        cr = __builtin_amdgcn_wmma_f32_16x16x32_f16(false, aH.v, false, bl, (short)0, cr, false, false);
        cr = __builtin_amdgcn_wmma_f32_16x16x32_f16(false, aL.v, false, bh, (short)0, cr, false, false);

        // ---- imag rows
        aH.h[0] = *reinterpret_cast<const v8h*>(sbH + rowI * FILTER_LEN + abase);
        aH.h[1] = *reinterpret_cast<const v8h*>(sbH + rowI * FILTER_LEN + abase + 16);
        aL.h[0] = *reinterpret_cast<const v8h*>(sbL + rowI * FILTER_LEN + abase);
        aL.h[1] = *reinterpret_cast<const v8h*>(sbL + rowI * FILTER_LEN + abase + 16);
        ci = __builtin_amdgcn_wmma_f32_16x16x32_f16(false, aH.v, false, bh, (short)0, ci, false, false);
        ci = __builtin_amdgcn_wmma_f32_16x16x32_f16(false, aH.v, false, bl, (short)0, ci, false, false);
        ci = __builtin_amdgcn_wmma_f32_16x16x32_f16(false, aL.v, false, bh, (short)0, ci, false, false);
    }

    // ---- magnitude + masked store (C/D layout: VGPR r -> M = r + 8*(lane>=16), N = lane&15)
    const int n = f0 + (lane & 15);
    if (n < NFRAMES) {
        #pragma unroll
        for (int r = 0; r < 8; ++r) {
            int m = mtile * 16 + r + ((lane >> 4) ? 8 : 0);
            if (m < CUTOFF) {
                float re = cr[r], im = ci[r];
                out[(long)m * NFRAMES + n] = sqrtf(re * re + im * im);
            }
        }
    }
}

extern "C" void kernel_launch(void* const* d_in, const int* in_sizes, int n_in,
                              void* d_out, int out_size, void* d_ws, size_t ws_size,
                              hipStream_t stream) {
    (void)in_sizes; (void)n_in; (void)out_size; (void)d_ws; (void)ws_size;
    const float* input = (const float*)d_in[0];   // (8000000,) f32
    const float* basis = (const float*)d_in[1];   // (802, 1, 800) f32
    float* out = (float*)d_out;                   // (1, 401, 40001) f32

    dim3 grid(MTILES, (NTILES + 3) / 4);          // 26 x 626
    dim3 block(128);                              // 4 waves (wave32)
    size_t shmem = 2u * 32u * FILTER_LEN * sizeof(_Float16);  // 100 KB basis tile
    stft_mag_wmma<<<grid, block, shmem, stream>>>(input, basis, out);
}
